// BahdanauAttention_76733885710447
// MI455X (gfx1250) — matrix operand order repaired and checked by reading, not compiled, not run on hardware
//
#include <hip/hip_runtime.h>
#include <hip/hip_bf16.h>

typedef __attribute__((ext_vector_type(16))) __bf16 v16bf;
typedef __attribute__((ext_vector_type(8)))  float  v8f;

union Frag16 { uint4 q[2]; v16bf v; };

__device__ __forceinline__ unsigned int f2bf(float x) {
    unsigned int u = __float_as_uint(x);
    return (u + 0x7FFFu + ((u >> 16) & 1u)) >> 16;   // RNE bf16
}
__device__ __forceinline__ unsigned int pack2bf(float a, float b) {
    return f2bf(a) | (f2bf(b) << 16);
}

// ---------------- constants ----------------
#define BATCH 32
#define SEQ   2048
#define HID   1024
#define DIM   2048        // 2H
#define MROWS 64          // rows per workgroup
#define ASTRIDE 40        // padded LDS row stride (elements) -> conflict-free ds_load_b128

// ---------------- kernel 1: Ua fp32 -> bf16 ----------------
__global__ __launch_bounds__(256) void ua_to_bf16(const float* __restrict__ Ua,
                                                  unsigned int* __restrict__ UaB) {
    int i = blockIdx.x * 256 + threadIdx.x;          // each handles 4 floats; 2M/4 = 512K threads
    const float4 f = ((const float4*)Ua)[i];
    uint2 p;
    p.x = pack2bf(f.x, f.y);
    p.y = pack2bf(f.z, f.w);
    ((uint2*)UaB)[i] = p;
}

// ---------------- kernel 2: h_proj = hidden @ Wa^T ----------------
__global__ __launch_bounds__(256) void hproj_kernel(const float* __restrict__ hidden,
                                                    const float* __restrict__ Wa,
                                                    float* __restrict__ hproj) {
    int g = blockIdx.x * 256 + threadIdx.x;          // 0..32767
    int b = g >> 10, h = g & 1023;
    const float4* hr = (const float4*)(hidden + (size_t)b * HID);
    const float4* wr = (const float4*)(Wa + (size_t)h * HID);
    float acc = 0.f;
    #pragma unroll 4
    for (int k = 0; k < HID / 4; ++k) {
        float4 a = hr[k], w = wr[k];
        acc += a.x * w.x + a.y * w.y + a.z * w.z + a.w * w.w;
    }
    hproj[g] = acc;
}

// ---------------- kernel 3: fused e_proj GEMM + relu + Va dot -> raw scores ----------------
// grid = 32 (b) * 32 (s-tile of 64 rows), block = 512 (16 waves).
// Each wave: 4 n-tiles x 4 m-tiles = 16 x v_wmma_f32_16x16x32_bf16 per K-step of 32.
// Steady-state loop is branch-free; final K-step peeled.
__global__ __launch_bounds__(512) void scores_wmma(const float* __restrict__ enc,        // [S,B,2H]
                                                   const unsigned short* __restrict__ UaB, // bf16 [H,2H]
                                                   const float* __restrict__ hproj,      // [B,H]
                                                   const float* __restrict__ Va,         // [H]
                                                   float* __restrict__ scores)           // [B,S]
{
    const int b     = blockIdx.x & 31;
    const int stile = blockIdx.x >> 5;              // 0..31
    const int row0  = stile * MROWS;
    const int tid   = threadIdx.x;
    const int lane  = tid & 31;
    const int wave  = tid >> 5;                     // 0..15

    __shared__ __align__(16) unsigned short sA[2][MROWS * ASTRIDE];  // double-buffered bf16 A tile
    __shared__ float sPart[16][MROWS];

    const v8f vzero = {0.f, 0.f, 0.f, 0.f, 0.f, 0.f, 0.f, 0.f};
    v8f acc[4][4];                                   // [n-tile][m-tile]
    #pragma unroll
    for (int t = 0; t < 4; ++t)
        #pragma unroll
        for (int mt = 0; mt < 4; ++mt) acc[t][mt] = vzero;

    const int l15    = lane & 15;
    const int kHalfA = (lane < 16) ? 0 : 8;   // A lane K groups: {kA..kA+7, kA+16..kA+23}
    const int kHalfB = (lane < 16) ? 0 : 8;   // B lane K groups: {kB..kB+7, kB+16..kB+23}

    // cooperative loader: each of 512 threads loads one float4 per K-step
    const int r  = tid >> 3;                  // row 0..63
    const int c4 = tid & 7;                   // float4 chunk 0..7
    const float* aptr = enc + ((size_t)(row0 + r) * BATCH + b) * DIM + c4 * 4;

    // prologue: stage K-step 0
    {
        const float4 f = *(const float4*)aptr;
        uint2 p; p.x = pack2bf(f.x, f.y); p.y = pack2bf(f.z, f.w);
        *(uint2*)&sA[0][r * ASTRIDE + c4 * 4] = p;
    }
    __syncthreads();

    // one K-step of fragment loads + 16 WMMAs
    auto gemm_step = [&](int kk, int cur) {
        Frag16 fa[4];
        #pragma unroll
        for (int mt = 0; mt < 4; ++mt) {
            const int base = (mt * 16 + l15) * ASTRIDE;
            fa[mt].q[0] = *(const uint4*)&sA[cur][base + kHalfA];
            fa[mt].q[1] = *(const uint4*)&sA[cur][base + kHalfA + 16];
        }
        Frag16 fb[4];
        #pragma unroll
        for (int t = 0; t < 4; ++t) {
            const unsigned short* pb = UaB + (size_t)((wave * 4 + t) * 16 + l15) * DIM + kk + kHalfB;
            fb[t].q[0] = ((const uint4*)pb)[0];
            fb[t].q[1] = ((const uint4*)pb)[2];
        }
        #pragma unroll
        for (int t = 0; t < 4; ++t)
            #pragma unroll
            for (int mt = 0; mt < 4; ++mt)
                acc[t][mt] = __builtin_amdgcn_wmma_f32_16x16x32_bf16(
                    false, fa[mt].v, false, fb[t].v, (short)0, acc[t][mt], false, false);
    };

    // steady state: branch-free, double-buffered
    for (int kk = 0; kk < DIM - 32; kk += 32) {
        const int cur = (kk >> 5) & 1;
        const float4 fn = *(const float4*)(aptr + kk + 32);   // next A tile, issued early
        #pragma unroll
        for (int t = 0; t < 4; ++t)                            // unconditional B prefetch, +2 steps
            __builtin_prefetch(UaB + (size_t)((wave * 4 + t) * 16 + l15) * DIM + kk + 64 + kHalfB, 0, 2);
        gemm_step(kk, cur);
        uint2 p; p.x = pack2bf(fn.x, fn.y); p.y = pack2bf(fn.z, fn.w);
        *(uint2*)&sA[cur ^ 1][r * ASTRIDE + c4 * 4] = p;
        __syncthreads();
    }
    // peeled final K-step
    gemm_step(DIM - 32, ((DIM / 32) - 1) & 1);

    // epilogue: score[m] = sum_n Va[n] * relu(acc + hproj[b][n])
    float p[4][8];
    #pragma unroll
    for (int mt = 0; mt < 4; ++mt)
        #pragma unroll
        for (int v = 0; v < 8; ++v) p[mt][v] = 0.f;

    #pragma unroll
    for (int t = 0; t < 4; ++t) {
        const int n  = (wave * 4 + t) * 16 + l15;
        const float hp = hproj[b * HID + n];
        const float va = Va[n];
        #pragma unroll
        for (int mt = 0; mt < 4; ++mt)
            #pragma unroll
            for (int v = 0; v < 8; ++v) {
                float e = acc[t][mt][v] + hp;
                e = e > 0.f ? e : 0.f;
                p[mt][v] += va * e;
            }
    }
    // reduce across the 16 lanes holding the n-dimension (per 16-lane half)
    #pragma unroll
    for (int mt = 0; mt < 4; ++mt)
        #pragma unroll
        for (int v = 0; v < 8; ++v)
            #pragma unroll
            for (int off = 1; off <= 8; off <<= 1)
                p[mt][v] += __shfl_xor(p[mt][v], off, 32);

    if (l15 == 0) {
        const int rbase = (lane < 16) ? 0 : 8;   // C layout: lanes 0-15 rows 0-7, 16-31 rows 8-15
        #pragma unroll
        for (int mt = 0; mt < 4; ++mt)
            #pragma unroll
            for (int v = 0; v < 8; ++v) sPart[wave][mt * 16 + rbase + v] = p[mt][v];
    }
    __syncthreads();
    if (tid < MROWS) {
        float s = 0.f;
        #pragma unroll
        for (int w = 0; w < 16; ++w) s += sPart[w][tid];
        scores[b * SEQ + row0 + tid] = s;
    }
}

// ---------------- kernel 4: softmax over S per batch ----------------
__global__ __launch_bounds__(256) void softmax_kernel(const float* __restrict__ scores,
                                                      float* __restrict__ attn) {
    const int b = blockIdx.x, tid = threadIdx.x;
    __shared__ float red[256];
    float x[8];
    float m = -3.402823466e+38f;
    #pragma unroll
    for (int i = 0; i < 8; ++i) {
        x[i] = scores[b * SEQ + tid + i * 256];
        m = fmaxf(m, x[i]);
    }
    red[tid] = m; __syncthreads();
    for (int s = 128; s > 0; s >>= 1) {
        if (tid < s) red[tid] = fmaxf(red[tid], red[tid + s]);
        __syncthreads();
    }
    m = red[0]; __syncthreads();
    float sum = 0.f;
    #pragma unroll
    for (int i = 0; i < 8; ++i) { x[i] = __expf(x[i] - m); sum += x[i]; }
    red[tid] = sum; __syncthreads();
    for (int s = 128; s > 0; s >>= 1) {
        if (tid < s) red[tid] += red[tid + s];
        __syncthreads();
    }
    const float inv = 1.f / red[0];
    #pragma unroll
    for (int i = 0; i < 8; ++i) attn[b * SEQ + tid + i * 256] = x[i] * inv;
}

// ---------------- kernel 5: context partial sums (segmented over S) ----------------
// grid = 8 seg * 2 chunk * 32 b = 512 blocks
__global__ __launch_bounds__(256) void context_partial(const float* __restrict__ enc,
                                                       const float* __restrict__ attn,
                                                       float4* __restrict__ part) {
    const int b     = blockIdx.x & 31;
    const int chunk = (blockIdx.x >> 5) & 1;
    const int seg   = blockIdx.x >> 6;               // 0..7
    const int d4    = chunk * 256 + threadIdx.x;     // float4 index 0..511
    float4 acc; acc.x = acc.y = acc.z = acc.w = 0.f;
    const int s0 = seg * 256;
    for (int s = s0; s < s0 + 256; ++s) {
        const float w = attn[b * SEQ + s];
        const float4 e = *(const float4*)(enc + ((size_t)s * BATCH + b) * DIM + (size_t)d4 * 4);
        acc.x += w * e.x; acc.y += w * e.y; acc.z += w * e.z; acc.w += w * e.w;
    }
    part[((size_t)seg * 32 + b) * 512 + d4] = acc;
}

// ---------------- kernel 6: reduce segments -> context in d_out ----------------
__global__ __launch_bounds__(256) void context_reduce(const float4* __restrict__ part,
                                                      float4* __restrict__ ctx) {
    const int i = blockIdx.x * 256 + threadIdx.x;    // 0..16383 ; b = i>>9, d4 = i&511
    float4 acc; acc.x = acc.y = acc.z = acc.w = 0.f;
    #pragma unroll
    for (int seg = 0; seg < 8; ++seg) {
        const float4 p = part[((size_t)seg * 32 + (i >> 9)) * 512 + (i & 511)];
        acc.x += p.x; acc.y += p.y; acc.z += p.z; acc.w += p.w;
    }
    ctx[i] = acc;
}

// ---------------- launch ----------------
extern "C" void kernel_launch(void* const* d_in, const int* in_sizes, int n_in,
                              void* d_out, int out_size, void* d_ws, size_t ws_size,
                              hipStream_t stream) {
    const float* hidden = (const float*)d_in[0];   // [32,1024]
    const float* enc    = (const float*)d_in[1];   // [2048,32,2048]
    const float* Wa     = (const float*)d_in[2];   // [1024,1024]
    const float* Ua     = (const float*)d_in[3];   // [1024,2048]
    const float* Va     = (const float*)d_in[4];   // [1,1024]

    float* out_ctx  = (float*)d_out;               // [32,1,2048] -> 65536 floats
    float* out_attn = (float*)d_out + 65536;       // [32,2048]   -> 65536 floats

    // workspace layout (bytes, 256-aligned)
    char* ws = (char*)d_ws;
    float*          hproj  = (float*)(ws + 0);                       // 128 KB
    float*          scores = (float*)(ws + 128 * 1024);              // 256 KB
    unsigned short* UaB    = (unsigned short*)(ws + 384 * 1024);     // 4 MB bf16
    float4*         part   = (float4*)(ws + (384 + 4096) * 1024);    // 2 MB

    ua_to_bf16     <<<2048, 256, 0, stream>>>(Ua, (unsigned int*)UaB);
    hproj_kernel   <<<128,  256, 0, stream>>>(hidden, Wa, hproj);
    scores_wmma    <<<1024, 512, 0, stream>>>(enc, UaB, hproj, Va, scores);
    softmax_kernel <<<32,   256, 0, stream>>>(scores, out_attn);
    context_partial<<<512,  256, 0, stream>>>(enc, out_attn, part);
    context_reduce <<<64,   256, 0, stream>>>(part, (float4*)out_ctx);
}
